// TransformerEncoderLayer_17179869294
// MI455X (gfx1250) — matrix operand order repaired and checked by reading, not compiled, not run on hardware
//
#include <hip/hip_runtime.h>
#include <hip/hip_bf16.h>

// ---------------------------------------------------------------------------
// Types for CDNA5 WMMA (wave32, 16x16x32 bf16 -> f32)
// ---------------------------------------------------------------------------
typedef __attribute__((ext_vector_type(16))) __bf16 v16bf;
typedef __attribute__((ext_vector_type(8)))  float  v8f;

union FragBF16 { v16bf v; unsigned u[8]; };
union AccF32   { v8f v; float f[8]; };

__device__ __forceinline__ unsigned short f2bf(float f) {
    unsigned u = __builtin_bit_cast(unsigned, f);
    unsigned r = u + 0x7FFFu + ((u >> 16) & 1u);   // round-to-nearest-even
    return (unsigned short)(r >> 16);
}

// CDNA5 async global->LDS copy (16B per lane), tracked by ASYNCcnt.
__device__ __forceinline__ void async_cp16(unsigned ldsOff, unsigned voff,
                                           unsigned long long base) {
    asm volatile("global_load_async_to_lds_b128 %0, %1, %2"
                 :: "v"(ldsOff), "v"(voff), "s"(base) : "memory");
}

// ---------------------------------------------------------------------------
// f32 -> bf16 with transpose: out[n*K + k] = bf16(in[k*N + n]).
// LDS-tiled 32x32, 256 threads (32x8).
// ---------------------------------------------------------------------------
__global__ __launch_bounds__(256)
void cvt_transpose_bf16(const float* __restrict__ in,
                        unsigned short* __restrict__ out, int K, int N) {
    __shared__ unsigned short tile[32][33];
    const int k0 = blockIdx.y * 32, n0 = blockIdx.x * 32;
    const int tx = threadIdx.x & 31, ty = threadIdx.x >> 5;
#pragma unroll
    for (int r = 0; r < 32; r += 8)
        tile[ty + r][tx] = f2bf(in[(size_t)(k0 + ty + r) * N + n0 + tx]);
    __syncthreads();
#pragma unroll
    for (int r = 0; r < 32; r += 8)
        out[(size_t)(n0 + ty + r) * K + k0 + tx] = tile[tx][ty + r];
}

// ---------------------------------------------------------------------------
// Fused LayerNorm (f32 in) -> bf16 out.  One 256-thread block per row of 1024.
// ---------------------------------------------------------------------------
__global__ __launch_bounds__(256)
void layernorm_bf16(const float* __restrict__ x, const float* __restrict__ g,
                    const float* __restrict__ b, unsigned short* __restrict__ out) {
    const int row = blockIdx.x;
    const float* xr = x + (size_t)row * 1024;
    const int t = threadIdx.x;
    float v[4];
    float s = 0.f, sq = 0.f;
#pragma unroll
    for (int i = 0; i < 4; ++i) {
        float f = xr[t + 256 * i];
        v[i] = f; s += f; sq += f * f;
    }
#pragma unroll
    for (int m = 16; m >= 1; m >>= 1) {
        s  += __shfl_xor(s, m, 32);
        sq += __shfl_xor(sq, m, 32);
    }
    __shared__ float ss[8], ssq[8];
    const int wave = t >> 5, lane = t & 31;
    if (lane == 0) { ss[wave] = s; ssq[wave] = sq; }
    __syncthreads();
    if (wave == 0) {
        float a  = (lane < 8) ? ss[lane]  : 0.f;
        float a2 = (lane < 8) ? ssq[lane] : 0.f;
#pragma unroll
        for (int m = 4; m >= 1; m >>= 1) {
            a  += __shfl_xor(a, m, 32);
            a2 += __shfl_xor(a2, m, 32);
        }
        if (lane == 0) { ss[0] = a; ssq[0] = a2; }
    }
    __syncthreads();
    const float mu   = ss[0] * (1.0f / 1024.0f);
    const float var  = ssq[0] * (1.0f / 1024.0f) - mu * mu;
    const float rstd = rsqrtf(var + 1e-5f);
#pragma unroll
    for (int i = 0; i < 4; ++i) {
        int c = t + 256 * i;
        out[(size_t)row * 1024 + c] = f2bf((v[i] - mu) * rstd * g[c] + b[c]);
    }
}

// ---------------------------------------------------------------------------
// bf16 GEMM, WMMA 16x16x32.  Block tile 128x128, K-step 32, 8 waves (2x4),
// each wave a 64x32 subtile (4x2 accumulators).  Tiles staged with
// GLOBAL_LOAD_ASYNC_TO_LDS_B128 into a double-buffered LDS (ASYNCcnt fenced).
// A: [M,K] bf16 row-major.  Wt: [N,K] bf16 (pre-transposed weights) so both
// tiles are row-contiguous 16B chunks in global memory.
// LDS rows padded to 40 bf16 (80B: 16B-aligned for async B128, 20-dword
// stride -> fragment-read lanes hit distinct banks).
// EPI: 0 = f32 out + residual,  1 = bf16 out + exact GELU,  2 = QKV scatter.
// ---------------------------------------------------------------------------
template <int EPI>
__global__ __launch_bounds__(256)
void gemm_bf16(const unsigned short* __restrict__ A,
               const unsigned short* __restrict__ Wt,
               const float* __restrict__ bias,
               void* __restrict__ out,
               const float* __restrict__ resid,
               int M, int N, int K,
               unsigned short* __restrict__ qOut,
               unsigned short* __restrict__ kOut,
               unsigned short* __restrict__ vOut) {
    __shared__ unsigned short lds[2][2][128 * 40];   // [buf][A=0/B=1][row][40]

    const int t    = threadIdx.x;
    const int lane = t & 31;
    const int wave = t >> 5;
    const int wr   = wave >> 2;      // 0..1
    const int wc   = wave & 3;       // 0..3
    const int l16  = lane & 15;
    const int lh   = lane >> 4;      // 0/1

    const int bM = blockIdx.y * 128;
    const int bN = blockIdx.x * 128;

    // Staging: tile = 128 rows x 32 bf16 = 512 x 16B chunks; 2 chunks/thread.
    // chunk c: linear = c*256 + t; row = linear/4; seg = linear%4.
    unsigned aVoff[2], ldsOffA[2][2], ldsOffB[2][2];
#pragma unroll
    for (int c = 0; c < 2; ++c) {
        const int linear = c * 256 + t;
        const int row = linear >> 2, seg = linear & 3;
        aVoff[c] = (unsigned)((row * K + seg * 8) * 2);          // bytes
#pragma unroll
        for (int bufi = 0; bufi < 2; ++bufi) {
            ldsOffA[bufi][c] =
                (unsigned)(size_t)&lds[bufi][0][0] + (unsigned)(row * 80 + seg * 16);
            ldsOffB[bufi][c] =
                (unsigned)(size_t)&lds[bufi][1][0] + (unsigned)(row * 80 + seg * 16);
        }
    }

    AccF32 acc[4][2];
#pragma unroll
    for (int i = 0; i < 4; ++i)
#pragma unroll
        for (int j = 0; j < 2; ++j)
#pragma unroll
            for (int r = 0; r < 8; ++r) acc[i][j].f[r] = 0.f;

    auto issue = [&](int bufi, int k0) {
        const unsigned long long abase =
            (unsigned long long)(size_t)(A + (size_t)bM * K + k0);
        const unsigned long long bbase =
            (unsigned long long)(size_t)(Wt + (size_t)bN * K + k0);
#pragma unroll
        for (int c = 0; c < 2; ++c) {
            async_cp16(ldsOffA[bufi][c], aVoff[c], abase);
            async_cp16(ldsOffB[bufi][c], aVoff[c], bbase);  // same row*K+seg layout
        }
    };

    issue(0, 0);
    int buf = 0;
    for (int k0 = 0; k0 < K; k0 += 32) {
        if (k0 + 32 < K) {
            issue(buf ^ 1, k0 + 32);
            asm volatile("s_wait_asynccnt 0x4" ::: "memory");  // current tile landed
        } else {
            asm volatile("s_wait_asynccnt 0x0" ::: "memory");
        }
        __syncthreads();

        const unsigned* Asu = (const unsigned*)&lds[buf][0][0];
        const unsigned* Bsu = (const unsigned*)&lds[buf][1][0];

        FragBF16 aF[4], bF[2];
#pragma unroll
        for (int i = 0; i < 4; ++i) {
            const int base = (wr * 64 + i * 16 + l16) * 20 + lh * 4;
#pragma unroll
            for (int j = 0; j < 4; ++j) {
                aF[i].u[j]     = Asu[base + j];       // K = lh*8 + 2j, +1
                aF[i].u[4 + j] = Asu[base + 8 + j];   // K = 16 + lh*8 + 2j, +1
            }
        }
#pragma unroll
        for (int j = 0; j < 2; ++j) {
            const int base = (wc * 32 + j * 16 + l16) * 20 + lh * 4;
#pragma unroll
            for (int v = 0; v < 8; ++v) bF[j].u[v] = Bsu[base + (v & 3) + (v >> 2) * 8]; // K = (v>>2)*16 + lh*8 + 2(v&3),+1
        }

#pragma unroll
        for (int i = 0; i < 4; ++i)
#pragma unroll
            for (int j = 0; j < 2; ++j)
                acc[i][j].v = __builtin_amdgcn_wmma_f32_16x16x32_bf16(
                    false, aF[i].v, false, bF[j].v, (short)0, acc[i][j].v,
                    false, false);

        __syncthreads();   // all waves done with lds[buf] before it is re-filled
        buf ^= 1;
    }

    // ---- epilogue ----
#pragma unroll
    for (int i = 0; i < 4; ++i)
#pragma unroll
        for (int j = 0; j < 2; ++j) {
            const int col = bN + wc * 32 + j * 16 + l16;
            const float bv = bias[col];
#pragma unroll
            for (int r = 0; r < 8; ++r) {
                const int row = bM + wr * 64 + i * 16 + lh * 8 + r;
                float v = acc[i][j].f[r] + bv;
                if constexpr (EPI == 0) {
                    ((float*)out)[(size_t)row * N + col] =
                        resid[(size_t)row * N + col] + v;
                } else if constexpr (EPI == 1) {
                    float gl = 0.5f * v * (1.0f + erff(v * 0.70710678118f));
                    ((unsigned short*)out)[(size_t)row * N + col] = f2bf(gl);
                } else {
                    const int part = col >> 10;
                    const int hh   = (col >> 6) & 15;
                    const int d    = col & 63;
                    const int bb   = row >> 10;
                    const int pos  = row & 1023;
                    const unsigned short bfv = f2bf(v);
                    if (part == 0)
                        qOut[((size_t)(bb * 16 + hh) * 1024 + pos) * 64 + d] = bfv;
                    else if (part == 1)
                        kOut[((size_t)(bb * 16 + hh) * 1024 + pos) * 64 + d] = bfv;
                    else  // V stored transposed [B,H,D,N] for contiguous PV B-frags
                        vOut[((size_t)(bb * 16 + hh) * 64 + d) * 1024 + pos] = bfv;
                }
            }
        }
}

// ---------------------------------------------------------------------------
// Flash attention: one wave per 16-query tile, 32-key chunks, online softmax.
// Q,K: [B,H,N,64] bf16;  V: [B,H,64,N] bf16;  O: [B,N,H*64] bf16.
// Grid: (qblocks=8, heads=16, batch=8), 256 threads = 8 waves.
// ---------------------------------------------------------------------------
__global__ __launch_bounds__(256)
void attn_kernel(const unsigned short* __restrict__ Q,
                 const unsigned short* __restrict__ Kb,
                 const unsigned short* __restrict__ Vb,
                 unsigned short* __restrict__ O) {
    __shared__ unsigned short pbuf[8][16 * 34];   // per-wave P staging, padded

    const int lane = threadIdx.x & 31;
    const int wave = threadIdx.x >> 5;
    const int l16  = lane & 15;
    const int lh   = lane >> 4;
    const int b    = blockIdx.z;
    const int h    = blockIdx.y;
    const int q0   = blockIdx.x * 128 + wave * 16;

    const unsigned short* qp = Q  + (size_t)(b * 16 + h) * 1024 * 64;
    const unsigned short* kp = Kb + (size_t)(b * 16 + h) * 1024 * 64;
    const unsigned short* vp = Vb + (size_t)(b * 16 + h) * 64 * 1024;

    FragBF16 aQ[2];
    {
        const unsigned* qu = (const unsigned*)(qp + (size_t)(q0 + l16) * 64);
#pragma unroll
        for (int ks = 0; ks < 2; ++ks)
#pragma unroll
            for (int j = 0; j < 8; ++j) {
                int k = (j < 4) ? (ks * 32 + lh * 8 + 2 * j)
                                : (ks * 32 + 16 + lh * 8 + 2 * (j - 4));
                aQ[ks].u[j] = qu[k >> 1];
            }
    }

    float mrow[8], lrow[8];
    AccF32 accO[4];
#pragma unroll
    for (int r = 0; r < 8; ++r) { mrow[r] = -3.0e38f; lrow[r] = 0.f; }
#pragma unroll
    for (int tt = 0; tt < 4; ++tt)
#pragma unroll
        for (int r = 0; r < 8; ++r) accO[tt].f[r] = 0.f;

    for (int kc = 0; kc < 1024; kc += 32) {
        AccF32 S[2];
#pragma unroll
        for (int jn = 0; jn < 2; ++jn)
#pragma unroll
            for (int r = 0; r < 8; ++r) S[jn].f[r] = 0.f;

#pragma unroll
        for (int ks = 0; ks < 2; ++ks) {
#pragma unroll
            for (int jn = 0; jn < 2; ++jn) {
                FragBF16 bK;
                const int key = kc + jn * 16 + l16;
                const unsigned* ku = (const unsigned*)(kp + (size_t)key * 64);
#pragma unroll
                for (int v = 0; v < 8; ++v) {
                    int d = ks * 32 + (v >> 2) * 16 + lh * 8 + 2 * (v & 3);
                    bK.u[v] = ku[d >> 1];
                }
                S[jn].v = __builtin_amdgcn_wmma_f32_16x16x32_bf16(
                    false, aQ[ks].v, false, bK.v, (short)0, S[jn].v, false, false);
            }
        }

        // ---- online softmax (scale = 1/sqrt(64) = 0.125) ----
        unsigned short* pb = pbuf[wave];
#pragma unroll
        for (int r = 0; r < 8; ++r) {
            float s0 = S[0].f[r] * 0.125f;
            float s1 = S[1].f[r] * 0.125f;
            float mx = fmaxf(s0, s1);
#pragma unroll
            for (int msk = 8; msk >= 1; msk >>= 1)
                mx = fmaxf(mx, __shfl_xor(mx, msk, 32));
            float mn = fmaxf(mrow[r], mx);
            float p0 = __expf(s0 - mn);
            float p1 = __expf(s1 - mn);
            float rs = p0 + p1;
#pragma unroll
            for (int msk = 8; msk >= 1; msk >>= 1)
                rs += __shfl_xor(rs, msk, 32);
            float alpha = __expf(mrow[r] - mn);
            lrow[r] = lrow[r] * alpha + rs;
            mrow[r] = mn;
#pragma unroll
            for (int tt = 0; tt < 4; ++tt) accO[tt].f[r] *= alpha;
            const int prow = r + lh * 8;
            pb[prow * 34 + l16]      = f2bf(p0);
            pb[prow * 34 + 16 + l16] = f2bf(p1);
        }

        // ---- reload P in A-matrix layout (per-wave LDS is in-order) ----
        FragBF16 aP;
        {
            const unsigned* pu = (const unsigned*)pb;
#pragma unroll
            for (int j = 0; j < 8; ++j) {
                int k = (j < 4) ? (lh * 8 + 2 * j) : (16 + lh * 8 + 2 * (j - 4));
                aP.u[j] = pu[l16 * 17 + (k >> 1)];
            }
        }

        // ---- O += P V  (V transposed: contiguous 32-bit loads) ----
#pragma unroll
        for (int tt = 0; tt < 4; ++tt) {
            FragBF16 bV;
            const int d = tt * 16 + l16;
            const unsigned* vu = (const unsigned*)(vp + (size_t)d * 1024 + kc);
#pragma unroll
            for (int v = 0; v < 8; ++v) bV.u[v] = vu[(v >> 2) * 8 + lh * 4 + (v & 3)];
            accO[tt].v = __builtin_amdgcn_wmma_f32_16x16x32_bf16(
                false, aP.v, false, bV.v, (short)0, accO[tt].v, false, false);
        }
    }

#pragma unroll
    for (int r = 0; r < 8; ++r) {
        const float inv = 1.0f / lrow[r];
        const int qrow = q0 + r + lh * 8;
#pragma unroll
        for (int tt = 0; tt < 4; ++tt) {
            const int d = tt * 16 + l16;
            O[((size_t)(b * 1024 + qrow)) * 1024 + h * 64 + d] =
                f2bf(accO[tt].f[r] * inv);
        }
    }
}

// ---------------------------------------------------------------------------
// Host launcher
// ---------------------------------------------------------------------------
extern "C" void kernel_launch(void* const* d_in, const int* in_sizes, int n_in,
                              void* d_out, int out_size, void* d_ws, size_t ws_size,
                              hipStream_t stream) {
    (void)in_sizes; (void)n_in; (void)out_size; (void)ws_size;

    const float* x      = (const float*)d_in[0];
    const float* ln1_g  = (const float*)d_in[1];
    const float* ln1_b  = (const float*)d_in[2];
    const float* qkv_w  = (const float*)d_in[3];
    const float* qkv_b  = (const float*)d_in[4];
    const float* proj_w = (const float*)d_in[5];
    const float* proj_b = (const float*)d_in[6];
    const float* ln2_g  = (const float*)d_in[7];
    const float* ln2_b  = (const float*)d_in[8];
    const float* fc1_w  = (const float*)d_in[9];
    const float* fc1_b  = (const float*)d_in[10];
    const float* fc2_w  = (const float*)d_in[11];
    const float* fc2_b  = (const float*)d_in[12];
    float* out = (float*)d_out;

    char* ws = (char*)d_ws;
    auto alloc = [&](size_t bytes) -> char* {
        char* p = ws;
        ws += (bytes + 255) & ~(size_t)255;
        return p;
    };

    const size_t T = 8 * 1024;   // tokens
    // Weights stored TRANSPOSED: [N][K] bf16
    unsigned short* qkvWt = (unsigned short*)alloc((size_t)3072 * 1024 * 2);
    unsigned short* projWt= (unsigned short*)alloc((size_t)1024 * 1024 * 2);
    unsigned short* fc1Wt = (unsigned short*)alloc((size_t)4096 * 1024 * 2);
    unsigned short* fc2Wt = (unsigned short*)alloc((size_t)1024 * 4096 * 2);
    unsigned short* hBuf  = (unsigned short*)alloc(T * 1024 * 2);
    unsigned short* qBuf  = (unsigned short*)alloc(T * 1024 * 2);
    unsigned short* kBuf  = (unsigned short*)alloc(T * 1024 * 2);
    unsigned short* vBuf  = (unsigned short*)alloc(T * 1024 * 2);
    unsigned short* oBuf  = (unsigned short*)alloc(T * 1024 * 2);
    float*          x1    = (float*)alloc(T * 1024 * 4);
    unsigned short* h2Buf = (unsigned short*)alloc(T * 1024 * 2);
    unsigned short* a1Buf = (unsigned short*)alloc(T * 4096 * 2);

    // 1) weights -> bf16, transposed to [N][K]
    cvt_transpose_bf16<<<dim3(3072 / 32, 1024 / 32), 256, 0, stream>>>(qkv_w, qkvWt, 1024, 3072);
    cvt_transpose_bf16<<<dim3(1024 / 32, 1024 / 32), 256, 0, stream>>>(proj_w, projWt, 1024, 1024);
    cvt_transpose_bf16<<<dim3(4096 / 32, 1024 / 32), 256, 0, stream>>>(fc1_w, fc1Wt, 1024, 4096);
    cvt_transpose_bf16<<<dim3(1024 / 32, 4096 / 32), 256, 0, stream>>>(fc2_w, fc2Wt, 4096, 1024);

    // 2) LN1 -> h (bf16)
    layernorm_bf16<<<8192, 256, 0, stream>>>(x, ln1_g, ln1_b, hBuf);

    // 3) QKV GEMM: h @ qkv_w + b  -> Q,K [B,H,N,64], V [B,H,64,N]
    gemm_bf16<2><<<dim3(3072 / 128, 8192 / 128), 256, 0, stream>>>(
        hBuf, qkvWt, qkv_b, nullptr, nullptr, 8192, 3072, 1024, qBuf, kBuf, vBuf);

    // 4) attention -> o (bf16, [B,N,C])
    attn_kernel<<<dim3(8, 16, 8), 256, 0, stream>>>(qBuf, kBuf, vBuf, oBuf);

    // 5) proj GEMM with residual: x1 = x + o @ proj_w + proj_b   (f32)
    gemm_bf16<0><<<dim3(1024 / 128, 8192 / 128), 256, 0, stream>>>(
        oBuf, projWt, proj_b, x1, x, 8192, 1024, 1024, nullptr, nullptr, nullptr);

    // 6) LN2 -> h2 (bf16)
    layernorm_bf16<<<8192, 256, 0, stream>>>(x1, ln2_g, ln2_b, h2Buf);

    // 7) FC1 + exact GELU -> a1 (bf16)
    gemm_bf16<1><<<dim3(4096 / 128, 8192 / 128), 256, 0, stream>>>(
        h2Buf, fc1Wt, fc1_b, a1Buf, nullptr, 8192, 4096, 1024, nullptr, nullptr, nullptr);

    // 8) FC2 with residual -> out (f32)
    gemm_bf16<0><<<dim3(1024 / 128, 8192 / 128), 256, 0, stream>>>(
        a1Buf, fc2Wt, fc2_b, out, x1, 8192, 1024, 4096, nullptr, nullptr, nullptr);
}
